// EnKF_LSTM_70609262346575
// MI455X (gfx1250) — matrix operand order repaired and checked by reading, not compiled, not run on hardware
//
#include <hip/hip_runtime.h>
#include <hip/hip_bf16.h>
#include <math.h>

// Problem constants (from reference): LS,BS,T,DIN,NH,N = 2,64,16,128,256,32
#define LSN   2
#define BSZ   64
#define TT    16
#define DINN  128
#define NHH   256
#define G4    1024     // 4*NH
#define NE    16       // ensemble half (N/2)
#define NN    32       // ensemble N

typedef __attribute__((ext_vector_type(16))) __bf16 v16bf;
typedef __attribute__((ext_vector_type(8)))  __bf16 v8bf;
typedef __attribute__((ext_vector_type(8)))  float  v8f;

static __device__ __forceinline__ float sigf(float x) { return 1.0f / (1.0f + __expf(-x)); }

// A-operand (16x32 MxK, bf16): lane L(0..15): M=L, elems 0..15 = K(kk+half*16..).
// One contiguous 32B chunk per lane.
static __device__ __forceinline__ v16bf load_a16(const __bf16* row, int kk, int half) {
  return *reinterpret_cast<const v16bf*>(row + kk + half * 16);
}

// B-operand (32x16 KxN, bf16): lane holds col n=lane&15, K = kk + half*16 + 0..15,
// i.e. 32 contiguous bytes of the row-major W row (since B = W^T).
static __device__ __forceinline__ v16bf load_b16(const __bf16* colrow, int kk, int half) {
  return *reinterpret_cast<const v16bf*>(colrow + kk + half * 16);
}

static __device__ __forceinline__ v8f wmma_bf16(v16bf a, v16bf b, v8f c) {
  return __builtin_amdgcn_wmma_f32_16x16x32_bf16(false, a, false, b, (short)0, c, false, false);
}

// ---------------------------------------------------------------- kernel 1: f32 -> bf16 staging
__global__ __launch_bounds__(256) void k_convert(
    const float* Wih0, const float* Whh0, const float* Wih1, const float* Whh1,
    const float* x, const float* bi0, const float* bh0, const float* bi1, const float* bh1,
    __bf16* Wih0b, __bf16* Whh0b, __bf16* Wih1b, __bf16* Whh1b, __bf16* xb,
    float* bias0, float* bias1) {
  int i = blockIdx.x * blockDim.x + threadIdx.x;
  if (i < G4 * DINN) Wih0b[i] = (__bf16)Wih0[i];
  if (i < G4 * NHH) {
    Whh0b[i] = (__bf16)Whh0[i];
    Wih1b[i] = (__bf16)Wih1[i];
    Whh1b[i] = (__bf16)Whh1[i];
  }
  if (i < BSZ * TT * DINN) xb[i] = (__bf16)x[i];
  if (i < G4) { bias0[i] = bi0[i] + bh0[i]; bias1[i] = bi1[i] + bh1[i]; }
}

// ---------------------------------------------------------------- kernel 2: 2-layer LSTM, one WG per ensemble
__global__ __launch_bounds__(1024) void k_lstm(
    const float* enkf, const float* noise_h, const float* noise_c,
    const __bf16* xb, const __bf16* Wih0b, const __bf16* Whh0b,
    const __bf16* Wih1b, const __bf16* Whh1b,
    const float* bias0, const float* bias1,
    const float* qv, const float* ev,
    __bf16* h0seq_all, float* state) {
  __shared__ __bf16 hbf[BSZ * NHH];  // current h as bf16 (32KB)
  const int e = blockIdx.x;          // ensemble member 0..15
  const int tid = threadIdx.x;
  const int lane = tid & 31, wave = tid >> 5;
  const int laneN = lane & 15, laneH = lane >> 4;
  const float qa = fabsf(qv[0]);
  const float ea = fabsf(ev[0]);
  __bf16* hseq = h0seq_all + (size_t)e * TT * BSZ * NHH;  // layer-0 output sequence (per ensemble)

  for (int l = 0; l < LSN; ++l) {
    const __bf16* Wih = l ? Wih1b : Wih0b;
    const __bf16* Whh = l ? Whh1b : Whh0b;
    const float* bias = l ? bias1 : bias0;
    const int Kin = l ? NHH : DINN;

    // init h (LDS, bf16) and c (registers) from enkf_state[l,:,:,e] / [l,:,:,16+e]
    for (int idx = tid; idx < BSZ * NHH; idx += 1024)
      hbf[idx] = (__bf16)enkf[((size_t)(l * BSZ + (idx >> 8)) * NHH + (idx & 255)) * NN + e];
    float c8[2][8];
#pragma unroll
    for (int gi = 0; gi < 2; ++gi) {
      int grp = wave * 2 + gi;
      int mb = (grp & 3) * 16, hb = (grp >> 2) * 16;
#pragma unroll
      for (int r = 0; r < 8; ++r) {
        int m = mb + r + laneH * 8, n = hb + laneN;
        c8[gi][r] = enkf[((size_t)(l * BSZ + m) * NHH + n) * NN + 16 + e];
      }
    }
    __syncthreads();

    for (int t = 0; t < TT; ++t) {
      v8f acc[2][4];
#pragma unroll
      for (int gi = 0; gi < 2; ++gi) {
        int grp = wave * 2 + gi, hb = (grp >> 2) * 16;
#pragma unroll
        for (int q = 0; q < 4; ++q) {
          float bv = bias[q * NHH + hb + laneN];
          v8f z;
#pragma unroll
          for (int r = 0; r < 8; ++r) z[r] = bv;
          acc[gi][q] = z;
        }
      }
      // g += x_t @ Wih^T  (layer0: xb ; layer1: layer-0 h sequence)
      for (int kk = 0; kk < Kin; kk += 32) {
#pragma unroll
        for (int gi = 0; gi < 2; ++gi) {
          int grp = wave * 2 + gi, mb = (grp & 3) * 16, hb = (grp >> 2) * 16;
          int m = mb + laneN;
          const __bf16* arow = l ? (hseq + (size_t)(t * BSZ + m) * NHH)
                                 : (xb + (size_t)(m * TT + t) * DINN);
          v16bf a = load_a16(arow, kk, laneH);
#pragma unroll
          for (int q = 0; q < 4; ++q) {
            int col = q * NHH + hb + laneN;
            v16bf b = load_b16(Wih + (size_t)col * Kin, kk, laneH);
            acc[gi][q] = wmma_bf16(a, b, acc[gi][q]);
          }
        }
      }
      // g += h @ Whh^T
      for (int kk = 0; kk < NHH; kk += 32) {
#pragma unroll
        for (int gi = 0; gi < 2; ++gi) {
          int grp = wave * 2 + gi, mb = (grp & 3) * 16, hb = (grp >> 2) * 16;
          int m = mb + laneN;
          v16bf a = load_a16(hbf + (size_t)m * NHH, kk, laneH);
#pragma unroll
          for (int q = 0; q < 4; ++q) {
            int col = q * NHH + hb + laneN;
            v16bf b = load_b16(Whh + (size_t)col * NHH, kk, laneH);
            acc[gi][q] = wmma_bf16(a, b, acc[gi][q]);
          }
        }
      }
      // pointwise LSTM cell update (C/D layout: row = base + r + laneH*8, col = base + laneN)
      float hnew[2][8];
#pragma unroll
      for (int gi = 0; gi < 2; ++gi) {
#pragma unroll
        for (int r = 0; r < 8; ++r) {
          float ig = sigf(acc[gi][0][r]);
          float fg = sigf(acc[gi][1][r]);
          float gg = tanhf(acc[gi][2][r]);
          float og = sigf(acc[gi][3][r]);
          float c2 = fg * c8[gi][r] + ig * gg;
          c8[gi][r] = c2;
          hnew[gi][r] = og * tanhf(c2);
        }
      }
      __syncthreads();
#pragma unroll
      for (int gi = 0; gi < 2; ++gi) {
        int grp = wave * 2 + gi, mb = (grp & 3) * 16, hb = (grp >> 2) * 16;
#pragma unroll
        for (int r = 0; r < 8; ++r) {
          int m = mb + r + laneH * 8, n = hb + laneN;
          __bf16 hv = (__bf16)hnew[gi][r];
          hbf[m * NHH + n] = hv;
          if (l == 0) hseq[(size_t)(t * BSZ + m) * NHH + n] = hv;
        }
      }
      __syncthreads();
      if (t == TT - 1) {
        // state[l,b,n, e] = hT + |q|*noise_h ; state[l,b,n, 16+e] = cT + |e|*noise_c
#pragma unroll
        for (int gi = 0; gi < 2; ++gi) {
          int grp = wave * 2 + gi, mb = (grp & 3) * 16, hb = (grp >> 2) * 16;
#pragma unroll
          for (int r = 0; r < 8; ++r) {
            int m = mb + r + laneH * 8, n = hb + laneN;
            size_t base = (size_t)(l * BSZ + m) * NHH + n;
            state[base * NN + e]      = hnew[gi][r] + qa * noise_h[base * NE + e];
            state[base * NN + 16 + e] = c8[gi][r]   + ea * noise_c[base * NE + e];
          }
        }
      }
    }
    __syncthreads();
  }
}

// ---------------------------------------------------------------- kernel 3: hxi, HA, centered A (bf16)
__global__ __launch_bounds__(256) void k_stats(const float* state, const float* Hm,
                                               float* hxi, float* HA, __bf16* Abf) {
  __shared__ float red[NN];
  __shared__ float hxm;
  int lb = blockIdx.x;   // (l*64+b), 0..127
  int n = threadIdx.x;   // hidden unit 0..255
  if (n < NN) red[n] = 0.f;
  __syncthreads();
  float sv[NN];
  const float* sp = state + ((size_t)lb * NHH + n) * NN;
#pragma unroll
  for (int i = 0; i < NN; ++i) sv[i] = sp[i];
  float hm = Hm[n];
#pragma unroll
  for (int i = 0; i < NN; ++i) atomicAdd(&red[i], hm * sv[i]);
  __syncthreads();
  if (n == 0) {
    float s = 0.f;
    for (int i = 0; i < NN; ++i) s += red[i];
    hxm = s / NN;
  }
  __syncthreads();
  if (n < NN) { hxi[lb * NN + n] = red[n]; HA[lb * NN + n] = red[n] - hxm; }
  float mean = 0.f;
#pragma unroll
  for (int i = 0; i < NN; ++i) mean += sv[i];
  mean *= (1.0f / NN);
  __bf16* ap = Abf + ((size_t)lb * NHH + n) * NN;
#pragma unroll
  for (int i = 0; i < NN; ++i) ap[i] = (__bf16)(sv[i] - mean);
}

// ---------------------------------------------------------------- kernel 4: P = HA HA^T /31 + (r^2+1e-6) I
__global__ __launch_bounds__(1024) void k_pmat(const float* HA, const float* rv, float* P) {
  int idx = blockIdx.x * blockDim.x + threadIdx.x;
  if (idx >= LSN * BSZ * BSZ) return;
  int l = idx >> 12, b = (idx >> 6) & 63, c = idx & 63;
  const float* ha = HA + (l * BSZ + b) * NN;
  const float* hc = HA + (l * BSZ + c) * NN;
  float s = 0.f;
#pragma unroll
  for (int i = 0; i < NN; ++i) s += ha[i] * hc[i];
  s *= (1.0f / (NN - 1));
  if (b == c) s += rv[0] * rv[0] + 1e-6f;
  P[idx] = s;
}

// ---------------------------------------------------------------- kernel 5: Pinv via Gauss-Jordan (P is SPD)
__global__ __launch_bounds__(256) void k_inv(const float* P, float* Pinv) {
  __shared__ float M[BSZ][2 * BSZ];
  __shared__ float fac[BSZ];
  int l = blockIdx.x, tid = threadIdx.x;
  for (int idx = tid; idx < BSZ * BSZ; idx += 256) {
    int r = idx >> 6, c = idx & 63;
    M[r][c] = P[(l * BSZ + r) * BSZ + c];
    M[r][64 + c] = (r == c) ? 1.f : 0.f;
  }
  __syncthreads();
  for (int k = 0; k < BSZ; ++k) {
    float pv = 1.f / M[k][k];
    __syncthreads();
    for (int j = tid; j < 2 * BSZ; j += 256) M[k][j] *= pv;
    __syncthreads();
    for (int i = tid; i < BSZ; i += 256) fac[i] = M[i][k];
    __syncthreads();
    for (int idx = tid; idx < BSZ * 2 * BSZ; idx += 256) {
      int i = idx >> 7, j = idx & 127;
      if (i != k) M[i][j] -= fac[i] * M[k][j];
    }
    __syncthreads();
  }
  for (int idx = tid; idx < BSZ * BSZ; idx += 256) {
    int r = idx >> 6, c = idx & 63;
    Pinv[(l * BSZ + r) * BSZ + c] = M[r][64 + c];
  }
}

// ---------------------------------------------------------------- kernel 6: S[l] = HA^T (Pinv innov) / 31 (stored transposed, bf16)
__global__ __launch_bounds__(1024) void k_smat(const float* Pinv, const float* hxi,
                                               const float* HA, const float* y, __bf16* STbf) {
  __shared__ float Ml[LSN * BSZ * NN];  // M[l,c,j] = sum_d Pinv[l,c,d]*(y[d]-hxi[1,d,j])
  int tid = threadIdx.x;
  for (int idx = tid; idx < LSN * BSZ * NN; idx += 1024) {
    int l = idx >> 11, c = (idx >> 5) & 63, j = idx & 31;
    const float* pr = Pinv + (l * BSZ + c) * BSZ;
    float s = 0.f;
    for (int d = 0; d < BSZ; ++d) s += pr[d] * (y[d] - hxi[(BSZ + d) * NN + j]);
    Ml[idx] = s;
  }
  __syncthreads();
  for (int idx = tid; idx < LSN * NN * NN; idx += 1024) {
    int l = idx >> 10, i = (idx >> 5) & 31, j = idx & 31;
    float s = 0.f;
    for (int c = 0; c < BSZ; ++c) s += HA[(l * BSZ + c) * NN + i] * Ml[(l * BSZ + c) * NN + j];
    STbf[(l * NN + j) * NN + i] = (__bf16)(s * (1.0f / (NN - 1)));  // [l][j][i]
  }
}

// ---------------------------------------------------------------- kernel 7: state_new = state + A @ S  (WMMA, K=32)
__global__ __launch_bounds__(256) void k_update(const __bf16* Abf, const __bf16* STbf,
                                                const float* state, float* state_new) {
  int lb = blockIdx.x;  // (l*64+b)
  int l = lb >> 6;
  int tid = threadIdx.x;
  int lane = tid & 31, wave = tid >> 5;
  int laneN = lane & 15, laneH = lane >> 4;
#pragma unroll
  for (int s = 0; s < 4; ++s) {
    int tile = wave * 4 + s;           // 32 tiles = 16 n-tiles x 2 j-tiles
    int nb = (tile >> 1) * 16, jb = (tile & 1) * 16;
    v16bf a = load_a16(Abf + ((size_t)lb * NHH + nb + laneN) * NN, 0, laneH);
    v16bf b = load_b16(STbf + (size_t)(l * NN + jb + laneN) * NN, 0, laneH);
    v8f c;
#pragma unroll
    for (int r = 0; r < 8; ++r)
      c[r] = state[((size_t)lb * NHH + nb + r + laneH * 8) * NN + jb + laneN];
    v8f d = wmma_bf16(a, b, c);
#pragma unroll
    for (int r = 0; r < 8; ++r)
      state_new[((size_t)lb * NHH + nb + r + laneH * 8) * NN + jb + laneN] = d[r];
  }
}

// ---------------------------------------------------------------- kernel 8: Y, Y_mu, sig
__global__ __launch_bounds__(1024) void k_ysig(const float* state_new, const float* Hm,
                                               const float* rv, float* out_ymu, float* out_sig) {
  __shared__ float Y[BSZ * NN];
  __shared__ float Ymu[BSZ];
  int tid = threadIdx.x;
  for (int idx = tid; idx < BSZ * NN; idx += 1024) {
    int b = idx >> 5, i = idx & 31;
    const float* sp = state_new + ((size_t)(BSZ + b) * NHH) * NN + i;  // layer 1
    float s = 0.f;
    for (int n = 0; n < NHH; ++n) s += sp[(size_t)n * NN] * Hm[n];
    Y[idx] = s;
  }
  __syncthreads();
  if (tid < BSZ) {
    float s = 0.f;
    for (int i = 0; i < NN; ++i) s += Y[tid * NN + i];
    Ymu[tid] = s / NN;
    out_ymu[tid] = s / NN;
  }
  __syncthreads();
  float r2 = rv[0] * rv[0];
  for (int idx = tid; idx < BSZ * BSZ; idx += 1024) {
    int b = idx >> 6, c = idx & 63;
    float s = 0.f;
    for (int i = 0; i < NN; ++i) s += (Y[b * NN + i] - Ymu[b]) * (Y[c * NN + i] - Ymu[c]);
    s *= (1.0f / (NN - 1));
    if (b == c) s += r2;
    out_sig[idx] = s;
  }
}

// ---------------------------------------------------------------- kernel 9: Cholesky, log-det, ll, nis
__global__ __launch_bounds__(64) void k_chol(const float* sig, const float* y, const float* ymu,
                                             float* out_ll, float* out_nis) {
  __shared__ float M[BSZ][BSZ];
  __shared__ float inn[BSZ], z[BSZ], w[BSZ];
  int tid = threadIdx.x;
  for (int idx = tid; idx < BSZ * BSZ; idx += 64) {
    int r = idx >> 6, c = idx & 63;
    float v = sig[idx];
    if (r == c) v += 1e-6f;
    M[r][c] = v;
  }
  if (tid < BSZ) inn[tid] = y[tid] - ymu[tid];
  __syncthreads();
  for (int k = 0; k < BSZ; ++k) {
    float d = sqrtf(M[k][k]);
    __syncthreads();
    if (tid == k) M[k][k] = d;
    else if (tid > k) M[tid][k] /= d;
    __syncthreads();
    if (tid > k) {
      float lik = M[tid][k];
      for (int j = k + 1; j <= tid; ++j) M[tid][j] -= lik * M[j][k];
    }
    __syncthreads();
  }
  if (tid == 0) {
    float logdet = 0.f;
    for (int k = 0; k < BSZ; ++k) logdet += 2.f * logf(M[k][k]);
    for (int i = 0; i < BSZ; ++i) {        // forward solve L z = inn
      float s = inn[i];
      for (int j = 0; j < i; ++j) s -= M[i][j] * z[j];
      z[i] = s / M[i][i];
    }
    for (int i = BSZ - 1; i >= 0; --i) {   // back solve L^T w = z
      float s = z[i];
      for (int j = i + 1; j < BSZ; ++j) s -= M[j][i] * w[j];
      w[i] = s / M[i][i];
    }
    float nis = 0.f;
    for (int i = 0; i < BSZ; ++i) nis += inn[i] * w[i];
    out_ll[0] = -0.5f * logdet - 0.5f * nis;
    out_nis[0] = nis;
  }
}

// ---------------------------------------------------------------- launch
extern "C" void kernel_launch(void* const* d_in, const int* in_sizes, int n_in,
                              void* d_out, int out_size, void* d_ws, size_t ws_size,
                              hipStream_t stream) {
  (void)in_sizes; (void)n_in; (void)out_size; (void)ws_size;
  const float* x       = (const float*)d_in[0];
  const float* y       = (const float*)d_in[1];
  const float* enkf    = (const float*)d_in[2];
  const float* noise_h = (const float*)d_in[3];
  const float* noise_c = (const float*)d_in[4];
  const float* Wih0    = (const float*)d_in[5];
  const float* Whh0    = (const float*)d_in[6];
  const float* bi0     = (const float*)d_in[7];
  const float* bh0     = (const float*)d_in[8];
  const float* Wih1    = (const float*)d_in[9];
  const float* Whh1    = (const float*)d_in[10];
  const float* bi1     = (const float*)d_in[11];
  const float* bh1     = (const float*)d_in[12];
  const float* Hm      = (const float*)d_in[13];
  const float* qv      = (const float*)d_in[14];
  const float* ev      = (const float*)d_in[15];
  const float* rv      = (const float*)d_in[16];

  char* ws = (char*)d_ws;
  size_t off = 0;
  auto alloc = [&](size_t bytes) -> char* {
    char* p = ws + off;
    off += (bytes + 255) & ~(size_t)255;
    return p;
  };
  __bf16* Wih0b = (__bf16*)alloc((size_t)G4 * DINN * 2);
  __bf16* Whh0b = (__bf16*)alloc((size_t)G4 * NHH * 2);
  __bf16* Wih1b = (__bf16*)alloc((size_t)G4 * NHH * 2);
  __bf16* Whh1b = (__bf16*)alloc((size_t)G4 * NHH * 2);
  __bf16* xb    = (__bf16*)alloc((size_t)BSZ * TT * DINN * 2);
  float*  bias0 = (float*)alloc((size_t)G4 * 4);
  float*  bias1 = (float*)alloc((size_t)G4 * 4);
  __bf16* h0seq = (__bf16*)alloc((size_t)NE * TT * BSZ * NHH * 2);
  float*  state = (float*)alloc((size_t)LSN * BSZ * NHH * NN * 4);
  float*  hxi   = (float*)alloc((size_t)LSN * BSZ * NN * 4);
  float*  HA    = (float*)alloc((size_t)LSN * BSZ * NN * 4);
  __bf16* Abf   = (__bf16*)alloc((size_t)LSN * BSZ * NHH * NN * 2);
  float*  P     = (float*)alloc((size_t)LSN * BSZ * BSZ * 4);
  float*  Pinv  = (float*)alloc((size_t)LSN * BSZ * BSZ * 4);
  __bf16* STbf  = (__bf16*)alloc((size_t)LSN * NN * NN * 2);

  float* out = (float*)d_out;
  const size_t OFF_SIG = BSZ;                                    // 64
  const size_t OFF_ST  = OFF_SIG + (size_t)BSZ * BSZ;            // 4160
  const size_t OFF_LL  = OFF_ST + (size_t)LSN * BSZ * NHH * NN;  // 1052736
  const size_t OFF_NIS = OFF_LL + 1;

  k_convert<<<1024, 256, 0, stream>>>(Wih0, Whh0, Wih1, Whh1, x, bi0, bh0, bi1, bh1,
                                      Wih0b, Whh0b, Wih1b, Whh1b, xb, bias0, bias1);
  k_lstm<<<NE, 1024, 0, stream>>>(enkf, noise_h, noise_c, xb, Wih0b, Whh0b, Wih1b, Whh1b,
                                  bias0, bias1, qv, ev, h0seq, state);
  k_stats<<<LSN * BSZ, 256, 0, stream>>>(state, Hm, hxi, HA, Abf);
  k_pmat<<<(LSN * BSZ * BSZ + 1023) / 1024, 1024, 0, stream>>>(HA, rv, P);
  k_inv<<<LSN, 256, 0, stream>>>(P, Pinv);
  k_smat<<<1, 1024, 0, stream>>>(Pinv, hxi, HA, y, STbf);
  k_update<<<LSN * BSZ, 256, 0, stream>>>(Abf, STbf, state, out + OFF_ST);
  k_ysig<<<1, 1024, 0, stream>>>(out + OFF_ST, Hm, rv, out, out + OFF_SIG);
  k_chol<<<1, 64, 0, stream>>>(out + OFF_SIG, y, out, out + OFF_LL, out + OFF_NIS);
}
